// WireframeDiscriminator_80453327389088
// MI455X (gfx1250) — matrix operand order repaired and checked by reading, not compiled, not run on hardware
//
#include <hip/hip_runtime.h>
#include <hip/hip_bf16.h>
#include <stdint.h>

// ---------------------------------------------------------------------------
// CDNA5 (gfx1250) implicit-GEMM implementation of the wireframe discriminator.
// Activations: NHWC f16, channels padded to a multiple of 32.
// Convs: V_WMMA_F32_16X16X32_F16, 4 waves/block = 4 adjacent Cout tiles.
// Weights are staged Global->LDS with GLOBAL_LOAD_ASYNC_TO_LDS_B128
// (ASYNCcnt-tracked DMA), double-buffered and synced with s_wait_asynccnt.
// BN uses batch statistics (training mode) -> explicit stats reduction pass.
// ---------------------------------------------------------------------------

typedef _Float16 v16h __attribute__((ext_vector_type(16)));
typedef _Float16 h8   __attribute__((ext_vector_type(8)));
typedef float    v8f  __attribute__((ext_vector_type(8)));

#define NEG_SLOPE 0.01f
#define BN_EPS    1e-5f

__device__ __forceinline__ float clamp01(float v) {
    return fminf(fmaxf(v, 0.0f), 1.0f);
}

// ---------------------------------------------------------------------------
// 1) Render: (64,46,18) -> NHWC f16 act0 (64,64,64,32)  [14 real ch + 18 pad]
// ---------------------------------------------------------------------------
__global__ __launch_bounds__(256)
void render_kernel(const float* __restrict__ elem, _Float16* __restrict__ act0) {
    __shared__ float se[46 * 18];
    const int b   = blockIdx.x >> 4;                         // 16 blocks/image
    const int pix = ((blockIdx.x & 15) << 8) + threadIdx.x;  // 0..4095
    for (int i = threadIdx.x; i < 46 * 18; i += 256)
        se[i] = elem[b * 46 * 18 + i];
    __syncthreads();

    const int h = pix >> 6, w = pix & 63;
    const float cy = (float)h, cx = (float)w;
    float acc[14];
#pragma unroll
    for (int c = 0; c < 14; ++c) acc[c] = 0.0f;

    for (int n = 0; n < 46; ++n) {
        const float* e = se + n * 18;
        const float x  = e[14] * 64.0f, y  = e[15] * 64.0f;
        const float ww = e[16] * 64.0f, hh = e[17] * 64.0f;
        const float x0 = x - 0.5f * ww, x1 = x + 0.5f * ww;
        const float y0 = y - 0.5f * hh, y1 = y + 0.5f * hh;
        const float by = clamp01(cy - y0) * clamp01(y1 - cy);
        const float bx = clamp01(cx - x0) * clamp01(x1 - cx);
        const float l0 = fmaxf(1.0f - fabsf(cx - x0), 0.0f) * by;
        const float l1 = fmaxf(1.0f - fabsf(cx - x1), 0.0f) * by;
        const float l2 = fmaxf(1.0f - fabsf(cy - y0), 0.0f) * bx;
        const float l3 = fmaxf(1.0f - fabsf(cy - y1), 0.0f) * bx;
        const float r  = fmaxf(fmaxf(l0, l1), fmaxf(l2, l3));
#pragma unroll
        for (int c = 0; c < 14; ++c) acc[c] = fmaf(e[c], r, acc[c]);
    }

    _Float16* o = act0 + ((size_t)b * 4096 + pix) * 32;
#pragma unroll
    for (int c = 0; c < 14; ++c) o[c] = (_Float16)acc[c];
#pragma unroll
    for (int c = 14; c < 32; ++c) o[c] = (_Float16)0.0f;
}

// ---------------------------------------------------------------------------
// 2) Weight packing: OIHW f32 -> per-(Ntile,tap,chunk) B-fragment blocks, f16.
// Block of 512 halfs; lane L's 16 halfs contiguous at L*16.
// B layout assumption (wave32, 32x16 KxN f16):
//   lanes 0-15  : column N=lane,    K 0-7 & 16-23
//   lanes 16-31 : column N=lane-16, K 8-15 & 24-31
// ---------------------------------------------------------------------------
__global__ __launch_bounds__(256)
void pack_weights_kernel(const float* __restrict__ w, _Float16* __restrict__ wp,
                         int taps, int chunks, int cinReal, int total) {
    const int idx = blockIdx.x * 256 + threadIdx.x;
    if (idx >= total) return;
    const int e    = idx & 15;
    const int lane = (idx >> 4) & 31;
    const int cc   = (idx >> 9) % chunks;
    const int t    = (idx / (512 * chunks)) % taps;
    const int nt   =  idx / (512 * chunks * taps);
    const int k    = cc * 32 + ((lane & 16) >> 1) + (e & 7) + ((e & 8) << 1);   // K index within CIN_PAD
    const int co   = nt * 16 + (lane & 15);
    float v = 0.0f;
    if (k < cinReal) v = w[((size_t)co * cinReal + k) * taps + t];
    wp[idx] = (_Float16)v;
}

// ---------------------------------------------------------------------------
// 3) Implicit-GEMM conv.
//    Block = 128 threads = 4 waves covering Cout tiles 4*blockIdx.y .. +3.
//    K loop: (KH*KW taps) x (CIN_PAD/32 chunks); each step = one
//    V_WMMA_F32_16X16X32_F16.
//    B fragments: staged Global->LDS via global_load_async_to_lds_b128,
//    double-buffered; each lane stages exactly the 32B it reads back, so the
//    only sync needed is s_wait_asynccnt (no workgroup barrier).
// A layout (16-bit A 16x32): lanes 0-15 row M=lane, K 0-7 & 16-23;
//                            lanes 16-31 row M=lane-16, K 8-15 & 24-31.
// C layout: VGPR j -> M = j + 8*(lane>=16), N = lane&15.
// ---------------------------------------------------------------------------
template<int H, int W, int OH, int OW, int CINP, int COUT, int KH, int KW,
         int ST, int PD>
__global__ __launch_bounds__(128)
void conv_wmma_kernel(const _Float16* __restrict__ in,
                      const _Float16* __restrict__ wp,
                      float* __restrict__ out) {
    constexpr int TAPS   = KH * KW;
    constexpr int CHUNKS = CINP / 32;
    constexpr int ITERS  = TAPS * CHUNKS;

    __shared__ __align__(16) _Float16 bbuf[2][4][512];   // double buf x 4 waves

    const int lane = threadIdx.x & 31;
    const int wave = threadIdx.x >> 5;
    const int tM   = blockIdx.x;
    const int tN   = blockIdx.y * 4 + wave;
    const int mrow = tM * 16 + (lane & 15);
    const int b    = mrow / (OH * OW);
    const int rem  = mrow % (OH * OW);
    const int oh   = rem / OW;
    const int ow   = rem % OW;
    const int sel  = lane >> 4;            // which K half-slice this lane owns

    // Base of this wave's B-fragment stream (lane-local 32B slices).
    const _Float16* wbase = wp + ((size_t)tN * ITERS) * 512 + lane * 16;
    // LDS byte offset of this lane's staging slot (low 32 bits of flat addr
    // of a __shared__ object == wave-relative LDS offset).
    const unsigned lds_lane =
        (unsigned)(uintptr_t)(&bbuf[0][wave][lane * 16]);
    constexpr unsigned BUF_STRIDE = 4u * 512u * 2u;      // bytes between bufs

    auto issueB = [&](int it, int buf) {
        const unsigned long long ga =
            (unsigned long long)(uintptr_t)(wbase + (size_t)it * 512);
        const unsigned lds = lds_lane + (unsigned)buf * BUF_STRIDE;
        asm volatile(
            "global_load_async_to_lds_b128 %0, %1, off\n\t"
            "global_load_async_to_lds_b128 %0, %1, off offset:16"
            :: "v"(lds), "v"(ga) : "memory");
    };

    v8f acc = {};
    issueB(0, 0);
    for (int it = 0; it < ITERS; ++it) {
        const int buf = it & 1;
        if (it + 1 < ITERS) {
            issueB(it + 1, buf ^ 1);
            // oldest 2 asyncs (current buffer) complete; next 2 in flight
            asm volatile("s_wait_asynccnt 0x2" ::: "memory");
        } else {
            asm volatile("s_wait_asynccnt 0x0" ::: "memory");
        }

        const int tap = it / CHUNKS;
        const int cc  = it % CHUNKS;
        const int kh  = tap / KW, kw = tap % KW;
        const int ih  = oh * ST - PD + kh;
        const int iw  = ow * ST - PD + kw;
        const bool inb = ((unsigned)ih < (unsigned)H) &&
                         ((unsigned)iw < (unsigned)W);

        h8 alo = {}, ahi = {};
        if (inb) {
            const _Float16* arow =
                in + (((size_t)b * H + ih) * W + iw) * CINP + cc * 32 + sel * 8;
            alo = *(const h8*)(arow);
            ahi = *(const h8*)(arow + 16);
        }
        v16h A = __builtin_shufflevector(
            alo, ahi, 0, 1, 2, 3, 4, 5, 6, 7, 8, 9, 10, 11, 12, 13, 14, 15);
        const v16h B = *(const v16h*)(&bbuf[buf][wave][lane * 16]);
        acc = __builtin_amdgcn_wmma_f32_16x16x32_f16(
            false, A, false, B, (short)0, acc, false, false);
    }

    const int n = tN * 16 + (lane & 15);
#pragma unroll
    for (int j = 0; j < 8; ++j) {
        const int m = tM * 16 + j + 8 * sel;
        out[(size_t)m * COUT + n] = acc[j];
    }
}

// ---------------------------------------------------------------------------
// 4) BN stats: one block per channel; stats[c] = sum, stats[C+c] = sumsq.
// ---------------------------------------------------------------------------
__global__ __launch_bounds__(256)
void bn_stats_kernel(const float* __restrict__ x, float* __restrict__ stats,
                     int M, int C) {
    __shared__ float s1[256], s2[256];
    const int c = blockIdx.x, tid = threadIdx.x;
    float a = 0.0f, b = 0.0f;
    for (int i = tid; i < M; i += 256) {
        const float v = x[(size_t)i * C + c];
        a += v;
        b += v * v;
    }
    s1[tid] = a; s2[tid] = b;
    __syncthreads();
    for (int s = 128; s > 0; s >>= 1) {
        if (tid < s) { s1[tid] += s1[tid + s]; s2[tid] += s2[tid + s]; }
        __syncthreads();
    }
    if (tid == 0) { stats[c] = s1[0]; stats[C + c] = s2[0]; }
}

// ---------------------------------------------------------------------------
// 5) BN normalize + LeakyReLU + f32->f16 (produces next layer's NHWC input).
// ---------------------------------------------------------------------------
__global__ __launch_bounds__(256)
void bn_lrelu_kernel(const float* __restrict__ x, const float* __restrict__ stats,
                     const float* __restrict__ g, const float* __restrict__ beta,
                     _Float16* __restrict__ y, int M, int C) {
    const int idx = blockIdx.x * 256 + threadIdx.x;
    if (idx >= M * C) return;
    const int c = idx % C;
    const float invM = 1.0f / (float)M;
    const float mean = stats[c] * invM;
    const float var  = stats[C + c] * invM - mean * mean;
    float v = (x[idx] - mean) * rsqrtf(var + BN_EPS) * g[c] + beta[c];
    v = (v >= 0.0f) ? v : NEG_SLOPE * v;
    y[idx] = (_Float16)v;
}

// ---------------------------------------------------------------------------
// 6) Head: 1x1 conv (1024 -> 1) + bias; one block per batch element.
// ---------------------------------------------------------------------------
__global__ __launch_bounds__(256)
void head_kernel(const _Float16* __restrict__ act4, const float* __restrict__ w5,
                 const float* __restrict__ b5, float* __restrict__ out) {
    __shared__ float s[256];
    const int b = blockIdx.x, tid = threadIdx.x;
    float a = 0.0f;
    for (int c = tid; c < 1024; c += 256)
        a += (float)act4[(size_t)b * 1024 + c] * w5[c];
    s[tid] = a;
    __syncthreads();
    for (int st = 128; st > 0; st >>= 1) {
        if (tid < st) s[tid] += s[tid + st];
        __syncthreads();
    }
    if (tid == 0) out[b] = s[0] + b5[0];
}

// ---------------------------------------------------------------------------
// Launch
// ---------------------------------------------------------------------------
extern "C" void kernel_launch(void* const* d_in, const int* in_sizes, int n_in,
                              void* d_out, int out_size, void* d_ws, size_t ws_size,
                              hipStream_t stream) {
    const float* input  = (const float*)d_in[0];
    const float* w1 = (const float*)d_in[2];
    const float* g1 = (const float*)d_in[3];
    const float* b1 = (const float*)d_in[4];
    const float* w2 = (const float*)d_in[5];
    const float* g2 = (const float*)d_in[6];
    const float* b2 = (const float*)d_in[7];
    const float* w3 = (const float*)d_in[8];
    const float* g3 = (const float*)d_in[9];
    const float* b3 = (const float*)d_in[10];
    const float* w4 = (const float*)d_in[11];
    const float* g4 = (const float*)d_in[12];
    const float* b4 = (const float*)d_in[13];
    const float* w5 = (const float*)d_in[14];
    const float* b5 = (const float*)d_in[15];
    float* out = (float*)d_out;

    char* ws = (char*)d_ws;
    // Workspace layout (256B-aligned offsets, bytes):
    _Float16* act0 = (_Float16*)(ws + 0);          // 64*64*64*32 f16 = 16 MiB
    float*    c1   = (float*)   (ws + 16777216);   // 64*32*32*64 f32 = 16 MiB
    _Float16* act1 = (_Float16*)(ws + 33554432);   // same count f16  =  8 MiB
    float*    c2   = (float*)   (ws + 41943040);   // 64*16*16*128    =  8 MiB
    _Float16* act2 = (_Float16*)(ws + 50331648);   //                  4 MiB
    float*    c3   = (float*)   (ws + 54525952);   // 64*8*8*256      =  4 MiB
    _Float16* act3 = (_Float16*)(ws + 58720256);   //                  2 MiB
    float*    c4   = (float*)   (ws + 60817408);   // 64*1024 f32
    _Float16* act4 = (_Float16*)(ws + 61079552);   // 64*1024 f16
    float*    stats= (float*)   (ws + 61210624);   // 2*1024 f32 (reused)
    _Float16* w1p  = (_Float16*)(ws + 61218816);   //  4*25*1*512 f16
    _Float16* w2p  = (_Float16*)(ws + 61321216);   //  8*25*2*512 f16
    _Float16* w3p  = (_Float16*)(ws + 61730816);   // 16*25*4*512 f16
    _Float16* w4p  = (_Float16*)(ws + 63369216);   // 64*64*8*512 f16 = 32 MiB

    // --- render ---
    render_kernel<<<1024, 256, 0, stream>>>(input, act0);

    // --- layer 1: 14(->32 pad) -> 64, 64x64 -> 32x32 ---
    {
        const int total = 4 * 25 * 1 * 512;
        pack_weights_kernel<<<(total + 255) / 256, 256, 0, stream>>>(w1, w1p, 25, 1, 14, total);
        conv_wmma_kernel<64, 64, 32, 32, 32, 64, 5, 5, 2, 2>
            <<<dim3(4096, 1), 128, 0, stream>>>(act0, w1p, c1);
        bn_stats_kernel<<<64, 256, 0, stream>>>(c1, stats, 65536, 64);
        bn_lrelu_kernel<<<(65536 * 64) / 256, 256, 0, stream>>>(c1, stats, g1, b1, act1, 65536, 64);
    }
    // --- layer 2: 64 -> 128, 32x32 -> 16x16 ---
    {
        const int total = 8 * 25 * 2 * 512;
        pack_weights_kernel<<<(total + 255) / 256, 256, 0, stream>>>(w2, w2p, 25, 2, 64, total);
        conv_wmma_kernel<32, 32, 16, 16, 64, 128, 5, 5, 2, 2>
            <<<dim3(1024, 2), 128, 0, stream>>>(act1, w2p, c2);
        bn_stats_kernel<<<128, 256, 0, stream>>>(c2, stats, 16384, 128);
        bn_lrelu_kernel<<<(16384 * 128) / 256, 256, 0, stream>>>(c2, stats, g2, b2, act2, 16384, 128);
    }
    // --- layer 3: 128 -> 256, 16x16 -> 8x8 ---
    {
        const int total = 16 * 25 * 4 * 512;
        pack_weights_kernel<<<(total + 255) / 256, 256, 0, stream>>>(w3, w3p, 25, 4, 128, total);
        conv_wmma_kernel<16, 16, 8, 8, 128, 256, 5, 5, 2, 2>
            <<<dim3(256, 4), 128, 0, stream>>>(act2, w3p, c3);
        bn_stats_kernel<<<256, 256, 0, stream>>>(c3, stats, 4096, 256);
        bn_lrelu_kernel<<<(4096 * 256) / 256, 256, 0, stream>>>(c3, stats, g3, b3, act3, 4096, 256);
    }
    // --- layer 4: 256 -> 1024, 8x8 valid -> 1x1 ---
    {
        const int total = 64 * 64 * 8 * 512;
        pack_weights_kernel<<<(total + 255) / 256, 256, 0, stream>>>(w4, w4p, 64, 8, 256, total);
        conv_wmma_kernel<8, 8, 1, 1, 256, 1024, 8, 8, 1, 0>
            <<<dim3(4, 16), 128, 0, stream>>>(act3, w4p, c4);
        bn_stats_kernel<<<1024, 256, 0, stream>>>(c4, stats, 64, 1024);
        bn_lrelu_kernel<<<(64 * 1024) / 256, 256, 0, stream>>>(c4, stats, g4, b4, act4, 64, 1024);
    }
    // --- head: 1x1 conv to scalar per batch element ---
    head_kernel<<<64, 256, 0, stream>>>(act4, w5, b5, out);

    (void)in_sizes; (void)n_in; (void)out_size; (void)ws_size;
}
